// SpikeAttention_49572512531138
// MI455X (gfx1250) — compile-verified, matrix operand order checked by reading, not run on hardware
//
#include <hip/hip_runtime.h>

// ---------------------------------------------------------------------------
// SpikeAttention on MI455X (gfx1250): bf16 WMMA pipeline, software-pipelined
// with copy-free ping-pong double buffering (K-loop unrolled by 2).
// amdgpu_waves_per_eu(1) on WMMA kernels: allow full VGPR file so the
// allocator doesn't shuffle accumulators to make room for load destinations.
//   hidden(f32) --cvt--> Xb(bf16)
//   Y = Xb @ W^T + b  via v_wmma_f32_16x16x32_bf16  (Q,K -> [BH,S,64]; V -> [BH,64,S])
//   flash attention per (b,h,16-query tile), online softmax, P staged via LDS,
//   V loads + next-K loads overlapped with the softmax VALU chain.
//   out = Ctx @ Wo^T + bo  (f32 store)
// ---------------------------------------------------------------------------

typedef __bf16 bf16_t;
typedef __attribute__((ext_vector_type(16))) __bf16 v16bf;
typedef __attribute__((ext_vector_type(8)))  __bf16 v8bf;
typedef __attribute__((ext_vector_type(8)))  float  v8f;

#define EMBED 2048
#define NHEAD 32
#define HDIM  64
#define SEQ   2048
#define BATCH 2
#define ROWS  (BATCH*SEQ)   // 4096 token rows

union FragU { v16bf v; v8bf h[2]; };

// Build a 16x32 bf16 A/B WMMA fragment from two contiguous 16B chunks.
// Per ISA: lane (m, c=lane>>4) holds K = {8c..8c+7} in VGPRs0-3 and
// K = {16+8c..16+8c+7} in VGPRs4-7 (relative to the fragment's K base).
static __device__ inline v16bf ldfrag(const bf16_t* lo, const bf16_t* hi) {
  FragU f;
  f.h[0] = *(const v8bf*)lo;
  f.h[1] = *(const v8bf*)hi;
  return f.v;
}

static __device__ inline v8f wmma_bf16(v16bf a, v16bf b, v8f c) {
  // (neg_a, A, neg_b, B, c_mod, C, reuse_a, reuse_b)
  return __builtin_amdgcn_wmma_f32_16x16x32_bf16(false, a, false, b, (short)0, c,
                                                 false, false);
}

// Ping-pong GEMM helpers: load one fragment set / consume one fragment set.
// Implicitly use xr0, xr1, wr[4], c, acc from the enclosing kernel scope.
#define LOAD_SET(A0, A1, BB, kbase)                                   \
  {                                                                   \
    const int lo_ = (kbase) + 8 * c, hi_ = (kbase) + 16 + 8 * c;      \
    A0 = ldfrag(xr0 + lo_, xr0 + hi_);                                \
    A1 = ldfrag(xr1 + lo_, xr1 + hi_);                                \
    BB[0] = ldfrag(wr[0] + lo_, wr[0] + hi_);                         \
    BB[1] = ldfrag(wr[1] + lo_, wr[1] + hi_);                         \
    BB[2] = ldfrag(wr[2] + lo_, wr[2] + hi_);                         \
    BB[3] = ldfrag(wr[3] + lo_, wr[3] + hi_);                         \
  }

#define MMA_SET(A0, A1, BB)                                           \
  {                                                                   \
    acc[0][0] = wmma_bf16(A0, BB[0], acc[0][0]);                      \
    acc[1][0] = wmma_bf16(A1, BB[0], acc[1][0]);                      \
    acc[0][1] = wmma_bf16(A0, BB[1], acc[0][1]);                      \
    acc[1][1] = wmma_bf16(A1, BB[1], acc[1][1]);                      \
    acc[0][2] = wmma_bf16(A0, BB[2], acc[0][2]);                      \
    acc[1][2] = wmma_bf16(A1, BB[2], acc[1][2]);                      \
    acc[0][3] = wmma_bf16(A0, BB[3], acc[0][3]);                      \
    acc[1][3] = wmma_bf16(A1, BB[3], acc[1][3]);                      \
  }

// ---------------------------------------------------------------------------
// Stage 1: f32 -> bf16 conversion, 8 elems/thread (b128 in, b128 out)
// ---------------------------------------------------------------------------
__global__ __launch_bounds__(256)
void cvt_f32_to_bf16(const float* __restrict__ src, bf16_t* __restrict__ dst, int n) {
  const int i = (blockIdx.x * 256 + threadIdx.x) * 8;
  if (i < n) {
    v8f f = *(const v8f*)(src + i);
    v8bf o;
#pragma unroll
    for (int j = 0; j < 8; ++j) o[j] = (bf16_t)f[j];
    *(v8bf*)(dst + i) = o;
  }
}

// ---------------------------------------------------------------------------
// Stage 2: projection GEMM  Y[r, n] = sum_k X[r,k] * W[n,k] + bias[n]
// One wave -> 32x64 output tile (8 WMMAs / 12 b128 loads per 32-k step),
// K-loop unrolled by 2 with ping-pong buffers (no register copies).
// mode 0/1: store bf16 at [b, h, s, d]   (Q, K)
// mode 2  : store bf16 at [b, h, d, s]   (V transposed, PV B-frags contiguous)
// ---------------------------------------------------------------------------
__global__ __launch_bounds__(128)
__attribute__((amdgpu_waves_per_eu(1)))
void proj_gemm_bf16(const bf16_t* __restrict__ X,   // [ROWS, EMBED]
                    const bf16_t* __restrict__ W,   // [EMBED, EMBED], row n = out feature
                    const float*  __restrict__ bias,
                    bf16_t* __restrict__ out, int mode) {
  const int wid  = blockIdx.x * 4 + (threadIdx.x >> 5);
  const int lane = threadIdx.x & 31;
  const int m    = lane & 15;
  const int c    = lane >> 4;
  const int tm   = wid >> 5;       // 0..127 : 32-row tile
  const int tn   = wid & 31;       // 0..31  : 64-col tile
  const int n0   = tn * 64;

  const bf16_t* xr0 = X + (size_t)(tm * 32 + m) * EMBED;
  const bf16_t* xr1 = xr0 + (size_t)16 * EMBED;
  const bf16_t* wr[4];
#pragma unroll
  for (int t = 0; t < 4; ++t)
    wr[t] = W + (size_t)(n0 + t * 16 + m) * EMBED;

  v8f z = {};
  v8f acc[2][4] = {{z, z, z, z}, {z, z, z, z}};

  v16bf a0A, a1A, bA[4];           // ping buffer
  v16bf a0B, a1B, bB[4];           // pong buffer

  LOAD_SET(a0A, a1A, bA, 0);
  for (int k0 = 0; k0 < EMBED - 64; k0 += 64) {
    LOAD_SET(a0B, a1B, bB, k0 + 32);
    MMA_SET(a0A, a1A, bA);
    LOAD_SET(a0A, a1A, bA, k0 + 64);
    MMA_SET(a0B, a1B, bB);
  }
  LOAD_SET(a0B, a1B, bB, EMBED - 32);
  MMA_SET(a0A, a1A, bA);
  MMA_SET(a0B, a1B, bB);

#pragma unroll
  for (int t = 0; t < 4; ++t) {
    const int col = n0 + t * 16 + m;          // this lane's N
    const float bv = bias[col];
    const int h = col >> 6, d = col & 63;
#pragma unroll
    for (int i = 0; i < 2; ++i) {
#pragma unroll
      for (int r = 0; r < 8; ++r) {
        const int gr = tm * 32 + i * 16 + r + 8 * c;   // global token row
        const int bb = gr >> 11, s = gr & (SEQ - 1);
        const float v = acc[i][t][r] + bv;
        if (mode == 2) { // V transposed: [b, h, d, s]
          out[(((size_t)(bb * NHEAD + h) * HDIM + d) * SEQ) + s] = (bf16_t)v;
        } else {         // Q / K: [b, h, s, d]
          out[(((size_t)(bb * NHEAD + h) * SEQ + s) * HDIM) + d] = (bf16_t)v;
        }
      }
    }
  }
}

// ---------------------------------------------------------------------------
// Stage 3: causal flash attention, one wave per (b*h, 16-query tile).
// scores = (Q K^T)/8 ; masked -> -10000 ; softmax(w/16): fold into ts = 1/128.
// After QK^T: issue V loads (this iter) + K loads (next iter), then run the
// softmax VALU chain to hide their latency. P relaid out via per-wave LDS.
// ---------------------------------------------------------------------------
__global__ __launch_bounds__(128)
__attribute__((amdgpu_waves_per_eu(1)))
void flash_attn_bf16(const bf16_t* __restrict__ Q,   // [B*H, S, 64]
                     const bf16_t* __restrict__ K,   // [B*H, S, 64]
                     const bf16_t* __restrict__ VT,  // [B*H, 64, S]
                     bf16_t* __restrict__ Ctx) {     // [B, S, EMBED]
  __shared__ bf16_t plds[4][16][32];                 // per-wave 16x32 P tile (4 KB)
  const int wslot = threadIdx.x >> 5;
  const int wid   = blockIdx.x * 4 + wslot;
  const int lane  = threadIdx.x & 31;
  const int m     = lane & 15;
  const int c     = lane >> 4;
  const int qt    = wid & 127;          // query tile (S/16 = 128)
  const int bh    = wid >> 7;           // 0..63

  const bf16_t* Qp = Q  + (size_t)bh * SEQ * HDIM;
  const bf16_t* Kp = K  + (size_t)bh * SEQ * HDIM;
  const bf16_t* Vp = VT + (size_t)bh * HDIM * SEQ;

  // Q fragment (16 queries x 64 hd), loaded once.
  const bf16_t* qr = Qp + (size_t)(qt * 16 + m) * HDIM;
  const v16bf qa0 = ldfrag(qr + 0  + 8 * c, qr + 16 + 8 * c);
  const v16bf qa1 = ldfrag(qr + 32 + 8 * c, qr + 48 + 8 * c);

  v8f z = {};
  v8f oacc[4] = {z, z, z, z};           // 16 queries x 64 hd, f32
  float rmax[8], rsum[8];
#pragma unroll
  for (int r = 0; r < 8; ++r) { rmax[r] = -1e30f; rsum[r] = 0.0f; }

  const float ts = 1.0f / 128.0f;       // (1/sqrt(64)) * (1/T_softmax)
  const int npair = (qt + 2) >> 1;      // key tiles processed in pairs of 16

  // preload K fragments for the first pair
  v16bf kc[2][2];                       // [sub-tile][k-half]
#pragma unroll
  for (int sub = 0; sub < 2; ++sub) {
    const bf16_t* kr = Kp + (size_t)(sub * 16 + m) * HDIM;
    kc[sub][0] = ldfrag(kr + 0  + 8 * c, kr + 16 + 8 * c);
    kc[sub][1] = ldfrag(kr + 32 + 8 * c, kr + 48 + 8 * c);
  }

  for (int kp = 0; kp < npair; ++kp) {
    const int k0 = kp * 32;

    // ---- scores: two 16x16 tiles via QK^T ----
    v8f sc[2];
#pragma unroll
    for (int sub = 0; sub < 2; ++sub) {
      v8f s = z;
      s = wmma_bf16(qa0, kc[sub][0], s);
      s = wmma_bf16(qa1, kc[sub][1], s);
      sc[sub] = s;
    }

    // ---- issue V loads (needed after softmax) and next-K loads now ----
    v16bf vb[4];
#pragma unroll
    for (int t = 0; t < 4; ++t) {
      const bf16_t* vr = Vp + (size_t)(t * 16 + m) * SEQ + k0;
      vb[t] = ldfrag(vr + 8 * c, vr + 16 + 8 * c);
    }
    if (kp + 1 < npair) {
      const int k1 = k0 + 32;
#pragma unroll
      for (int sub = 0; sub < 2; ++sub) {
        const bf16_t* kr = Kp + (size_t)(k1 + sub * 16 + m) * HDIM;
        kc[sub][0] = ldfrag(kr + 0  + 8 * c, kr + 16 + 8 * c);
        kc[sub][1] = ldfrag(kr + 32 + 8 * c, kr + 48 + 8 * c);
      }
    }

    // ---- mask + temperature scale + online softmax (hides load latency) ----
#pragma unroll
    for (int r = 0; r < 8; ++r) {
      const int qrow = qt * 16 + r + 8 * c;
#pragma unroll
      for (int sub = 0; sub < 2; ++sub) {
        const int key = k0 + sub * 16 + m;       // this lane's key column
        float t = sc[sub][r] * ts;
        if (key > qrow) t = -625.0f;             // -10000 / 16
        sc[sub][r] = t;
      }
      // row max across the 16 lanes owning this row (xor<16 stays in group)
      float v = fmaxf(sc[0][r], sc[1][r]);
      v = fmaxf(v, __shfl_xor(v, 1));
      v = fmaxf(v, __shfl_xor(v, 2));
      v = fmaxf(v, __shfl_xor(v, 4));
      v = fmaxf(v, __shfl_xor(v, 8));
      const float nm = fmaxf(rmax[r], v);

      const float p0 = __expf(sc[0][r] - nm);
      const float p1 = __expf(sc[1][r] - nm);
      float ps = p0 + p1;
      ps += __shfl_xor(ps, 1);
      ps += __shfl_xor(ps, 2);
      ps += __shfl_xor(ps, 4);
      ps += __shfl_xor(ps, 8);

      const float f = __expf(rmax[r] - nm);      // rescale factor (0 on first iter)
      rsum[r] = rsum[r] * f + ps;
      rmax[r] = nm;
      oacc[0][r] *= f; oacc[1][r] *= f; oacc[2][r] *= f; oacc[3][r] *= f;

      // C-layout -> LDS row-major P tile
      plds[wslot][r + 8 * c][0  + m] = (bf16_t)p0;
      plds[wslot][r + 8 * c][16 + m] = (bf16_t)p1;
    }

    asm volatile("s_wait_dscnt 0x0" ::: "memory");

    // ---- reload P as a 16x32 A fragment ----
    FragU pf;
    pf.h[0] = *(const v8bf*)&plds[wslot][m][8 * c];
    pf.h[1] = *(const v8bf*)&plds[wslot][m][16 + 8 * c];
    const v16bf pa = pf.v;

    // ---- o += P @ V  (V^T rows are contiguous in keys) ----
#pragma unroll
    for (int t = 0; t < 4; ++t)
      oacc[t] = wmma_bf16(pa, vb[t], oacc[t]);
  }

  // ---- normalize and store context bf16: [b, s, h*64 + d] ----
  const int b = bh >> 5, h = bh & 31;
#pragma unroll
  for (int r = 0; r < 8; ++r) {
    const float inv = 1.0f / rsum[r];
    const int s = qt * 16 + r + 8 * c;
    const size_t base = ((size_t)(b * SEQ + s)) * EMBED + h * HDIM;
#pragma unroll
    for (int t = 0; t < 4; ++t)
      Ctx[base + t * 16 + m] = (bf16_t)(oacc[t][r] * inv);
  }
}

// ---------------------------------------------------------------------------
// Stage 4: output projection, f32 store to d_out:  out = Ctx @ Wo^T + bo
// Same copy-free ping-pong structure as proj_gemm_bf16.
// ---------------------------------------------------------------------------
__global__ __launch_bounds__(128)
__attribute__((amdgpu_waves_per_eu(1)))
void out_gemm_f32(const bf16_t* __restrict__ X,   // [ROWS, EMBED]
                  const bf16_t* __restrict__ W,   // [EMBED, EMBED]
                  const float*  __restrict__ bias,
                  float* __restrict__ out) {
  const int wid  = blockIdx.x * 4 + (threadIdx.x >> 5);
  const int lane = threadIdx.x & 31;
  const int m    = lane & 15;
  const int c    = lane >> 4;
  const int tm   = wid >> 5;
  const int tn   = wid & 31;
  const int n0   = tn * 64;

  const bf16_t* xr0 = X + (size_t)(tm * 32 + m) * EMBED;
  const bf16_t* xr1 = xr0 + (size_t)16 * EMBED;
  const bf16_t* wr[4];
#pragma unroll
  for (int t = 0; t < 4; ++t)
    wr[t] = W + (size_t)(n0 + t * 16 + m) * EMBED;

  v8f z = {};
  v8f acc[2][4] = {{z, z, z, z}, {z, z, z, z}};

  v16bf a0A, a1A, bA[4];
  v16bf a0B, a1B, bB[4];

  LOAD_SET(a0A, a1A, bA, 0);
  for (int k0 = 0; k0 < EMBED - 64; k0 += 64) {
    LOAD_SET(a0B, a1B, bB, k0 + 32);
    MMA_SET(a0A, a1A, bA);
    LOAD_SET(a0A, a1A, bA, k0 + 64);
    MMA_SET(a0B, a1B, bB);
  }
  LOAD_SET(a0B, a1B, bB, EMBED - 32);
  MMA_SET(a0A, a1A, bA);
  MMA_SET(a0B, a1B, bB);

#pragma unroll
  for (int t = 0; t < 4; ++t) {
    const int col = n0 + t * 16 + m;
    const float bv = bias[col];
#pragma unroll
    for (int i = 0; i < 2; ++i) {
#pragma unroll
      for (int r = 0; r < 8; ++r) {
        const int gr = tm * 32 + i * 16 + r + 8 * c;
        out[(size_t)gr * EMBED + col] = acc[i][t][r] + bv;
      }
    }
  }
}

// ---------------------------------------------------------------------------
extern "C" void kernel_launch(void* const* d_in, const int* in_sizes, int n_in,
                              void* d_out, int out_size, void* d_ws, size_t ws_size,
                              hipStream_t stream) {
  const float* hidden = (const float*)d_in[0];
  const float* Wq = (const float*)d_in[1];
  const float* bq = (const float*)d_in[2];
  const float* Wk = (const float*)d_in[3];
  const float* bk = (const float*)d_in[4];
  const float* Wv = (const float*)d_in[5];
  const float* bv = (const float*)d_in[6];
  const float* Wo = (const float*)d_in[7];
  const float* bo = (const float*)d_in[8];
  float* out = (float*)d_out;

  char* ws = (char*)d_ws;
  const size_t szX = (size_t)ROWS * EMBED * sizeof(bf16_t);                 // 16 MB
  const size_t szW = (size_t)EMBED * EMBED * sizeof(bf16_t);                //  8 MB
  const size_t szH = (size_t)BATCH * NHEAD * SEQ * HDIM * sizeof(bf16_t);   // 16 MB
  bf16_t* Xb  = (bf16_t*)(ws);
  bf16_t* Wqb = (bf16_t*)(ws + szX);
  bf16_t* Wkb = (bf16_t*)(ws + szX + 1 * szW);
  bf16_t* Wvb = (bf16_t*)(ws + szX + 2 * szW);
  bf16_t* Wob = (bf16_t*)(ws + szX + 3 * szW);
  bf16_t* Qb  = (bf16_t*)(ws + szX + 4 * szW);
  bf16_t* Kb  = (bf16_t*)(ws + szX + 4 * szW + 1 * szH);
  bf16_t* VTb = (bf16_t*)(ws + szX + 4 * szW + 2 * szH);
  bf16_t* Cxb = (bf16_t*)(ws + szX + 4 * szW + 3 * szH);                    // 112 MB total

  const int nX = ROWS * EMBED;          // 8,388,608
  const int nW = EMBED * EMBED;         // 4,194,304
  cvt_f32_to_bf16<<<nX / (256 * 8), 256, 0, stream>>>(hidden, Xb, nX);
  cvt_f32_to_bf16<<<nW / (256 * 8), 256, 0, stream>>>(Wq, Wqb, nW);
  cvt_f32_to_bf16<<<nW / (256 * 8), 256, 0, stream>>>(Wk, Wkb, nW);
  cvt_f32_to_bf16<<<nW / (256 * 8), 256, 0, stream>>>(Wv, Wvb, nW);
  cvt_f32_to_bf16<<<nW / (256 * 8), 256, 0, stream>>>(Wo, Wob, nW);

  dim3 gg(1024), blk(128);              // 4096 waves = (128 M-tiles x 32 N-tiles)
  proj_gemm_bf16<<<gg, blk, 0, stream>>>(Xb, Wqb, bq, Qb, 0);
  proj_gemm_bf16<<<gg, blk, 0, stream>>>(Xb, Wkb, bk, Kb, 1);
  proj_gemm_bf16<<<gg, blk, 0, stream>>>(Xb, Wvb, bv, VTb, 2);

  dim3 ga(2048);                        // 8192 waves = 64 bh x 128 q-tiles
  flash_attn_bf16<<<ga, blk, 0, stream>>>(Qb, Kb, VTb, Cxb);

  out_gemm_f32<<<gg, blk, 0, stream>>>(Cxb, Wob, bo, out);
}
